// FusionMambaBlock_25649544692419
// MI455X (gfx1250) — matrix operand order repaired and checked by reading, not compiled, not run on hardware
//
#include <hip/hip_runtime.h>
#include <stdint.h>

// ---------------- types for CDNA5 WMMA / TDM ----------------
typedef __attribute__((ext_vector_type(16))) __bf16 v16bf;
typedef __attribute__((ext_vector_type(8)))  float  v8f;
typedef __attribute__((ext_vector_type(4)))  unsigned int u32x4;
typedef __attribute__((ext_vector_type(8)))  unsigned int u32x8;

#define DM     1024
#define DI     2048
#define DSTATE 16
#define DTR    64
#define LSEQ   1024
#define MROWS  4096   // 2 streams * B(2) * L(1024)

__device__ __forceinline__ unsigned short f2bf(float f) {
  unsigned int u = __float_as_uint(f);
  u += 0x7fffu + ((u >> 16) & 1u);          // round-to-nearest-even
  return (unsigned short)(u >> 16);
}
__device__ __forceinline__ float siluf(float v) { return v / (1.f + __expf(-v)); }

// 16B global -> LDS, asynchronous (ASYNCcnt). ldsOff = LDS byte address.
__device__ __forceinline__ void async_g2l_b128(unsigned ldsOff, const void* g) {
  unsigned long long ga = (unsigned long long)(uintptr_t)g;
  asm volatile("global_load_async_to_lds_b128 %0, %1, off"
               :: "v"(ldsOff), "v"(ga) : "memory");
}
__device__ __forceinline__ void wait_asynccnt0() {
  asm volatile("s_wait_asynccnt 0x0" ::: "memory");
}

// =====================================================================
// Generic bf16 WMMA GEMM:  C[M,N](f32) = A[M,K](bf16) * Bt[N,K](bf16)^T
// Block tile 128x128, K-step 32, 8 waves (wave32), each wave 32x64 via
// a 2x4 grid of v_wmma_f32_16x16x32_bf16. Double-buffered LDS filled by
// global_load_async_to_lds_b128 (ASYNCcnt) overlapped with WMMA.
// =====================================================================
#define TM 128
#define TN 128
#define TK 32
#define LDSK 40   // padded LDS row stride in halfs (80B, 16B-aligned)
#define ATILE (TM * LDSK)
#define BTILE (TN * LDSK)

union Frag { uint4 u[2]; v16bf v; };

__global__ __launch_bounds__(256) void gemm_bf16_wmma(
    const unsigned short* __restrict__ A, int lda,
    const unsigned short* __restrict__ Bt, int ldb,
    float* __restrict__ C, int ldc, int M, int N, int K)
{
  __shared__ unsigned short As[2 * ATILE];
  __shared__ unsigned short Bs[2 * BTILE];

  const int tid  = threadIdx.x;
  const int lane = tid & 31;
  const int wave = tid >> 5;
  const int wm = wave & 3;        // 4 waves along M
  const int wn = wave >> 2;       // 2 waves along N
  const int m0 = blockIdx.y * TM;
  const int n0 = blockIdx.x * TN;

  // cooperative tile loader: each thread moves 32B per matrix per K-step
  const int lr = tid >> 1;              // row in tile 0..127
  const int lk = (tid & 1) * 16;        // half-offset in K: 0/16

  // fragment lane decomposition
  const int fm = lane & 15;             // row (A) / col (B/C)
  const int fh = lane >> 4;             // half-wave select

  // LDS byte addresses (low 32 bits of the LDS aperture = LDS offset)
  const unsigned asAddr = (unsigned)(uintptr_t)&As[0];
  const unsigned bsAddr = (unsigned)(uintptr_t)&Bs[0];
  const unsigned ldOff  = (unsigned)(lr * LDSK + lk) * 2u;

  const int ar = m0 + lr;
  const int br = n0 + lr;
  const bool aIn = ar < M;
  const bool bIn = br < N;
  const uint4 z4 = {0u, 0u, 0u, 0u};

  // stage K-tile `kt` into buffer `buf`
  auto stage = [&](int kt, int buf) {
    const unsigned aoff = asAddr + (unsigned)buf * (ATILE * 2u) + ldOff;
    const unsigned boff = bsAddr + (unsigned)buf * (BTILE * 2u) + ldOff;
    if (aIn) {
      const unsigned short* g = A + (size_t)ar * lda + kt + lk;
      async_g2l_b128(aoff, g);
      async_g2l_b128(aoff + 16u, g + 8);
    } else {
      *(uint4*)&As[buf * ATILE + lr * LDSK + lk]     = z4;
      *(uint4*)&As[buf * ATILE + lr * LDSK + lk + 8] = z4;
    }
    if (bIn) {
      const unsigned short* g = Bt + (size_t)br * ldb + kt + lk;
      async_g2l_b128(boff, g);
      async_g2l_b128(boff + 16u, g + 8);
    } else {
      *(uint4*)&Bs[buf * BTILE + lr * LDSK + lk]     = z4;
      *(uint4*)&Bs[buf * BTILE + lr * LDSK + lk + 8] = z4;
    }
  };

  v8f acc[2][4] = {};

  stage(0, 0);
  wait_asynccnt0();
  __syncthreads();

  const int nk = K / TK;
  for (int ki = 0; ki < nk; ++ki) {
    const int buf = ki & 1;
    if (ki + 1 < nk) stage((ki + 1) * TK, buf ^ 1);   // DMA next tile

    const unsigned short* as = As + buf * ATILE;
    const unsigned short* bs = Bs + buf * BTILE;

    Frag af[2], bfr[4];
    #pragma unroll
    for (int mi = 0; mi < 2; ++mi) {
      // A 16-bit 16x32 layout: lane<16 -> K {0..7,16..23}; lane>=16 -> K {8..15,24..31}
      const int base = (wm * 32 + mi * 16 + fm) * LDSK;
      af[mi].u[0] = *(const uint4*)&as[base + fh * 8];
      af[mi].u[1] = *(const uint4*)&as[base + 16 + fh * 8];
    }
    #pragma unroll
    for (int ni = 0; ni < 4; ++ni) {
      // B 16-bit 32x16 layout: lane<16 -> K {0..7,16..23}; lane>=16 -> K {8..15,24..31}
      const int base = (wn * 64 + ni * 16 + fm) * LDSK;
      bfr[ni].u[0] = *(const uint4*)&bs[base + fh * 8];
      bfr[ni].u[1] = *(const uint4*)&bs[base + 16 + fh * 8];
    }
    #pragma unroll
    for (int mi = 0; mi < 2; ++mi)
      #pragma unroll
      for (int ni = 0; ni < 4; ++ni)
        acc[mi][ni] = __builtin_amdgcn_wmma_f32_16x16x32_bf16(
            false, af[mi].v, false, bfr[ni].v,
            (short)0, acc[mi][ni], false, false);

    wait_asynccnt0();     // next tile fully in LDS
    __syncthreads();      // visible to all waves; this tile's reads done
  }

  // C/D 16x16 f32 layout: VGPR r, lanes0-15 -> M=r, lanes16-31 -> M=8+r; N=lane%16
  #pragma unroll
  for (int mi = 0; mi < 2; ++mi) {
    #pragma unroll
    for (int ni = 0; ni < 4; ++ni) {
      const int col = n0 + wn * 64 + ni * 16 + fm;
      if (col >= N) continue;
      const int rbase = m0 + wm * 32 + mi * 16 + fh * 8;
      #pragma unroll
      for (int r = 0; r < 8; ++r) {
        const int row = rbase + r;
        if (row < M) C[(size_t)row * ldc + col] = acc[mi][ni][r];
      }
    }
  }
}

// ---------------- elementwise / helper kernels ----------------

__global__ __launch_bounds__(256) void cvt_f32_bf16_k(
    const float* __restrict__ src, unsigned short* __restrict__ dst, int n)
{
  int i = blockIdx.x * 256 + threadIdx.x;
  if (i < n) dst[i] = f2bf(src[i]);
}

// transpose-convert: src f32 [Kr,Nc] row-major -> dst bf16 [Nc,Kr] row-major
__global__ __launch_bounds__(256) void tcvt_k(
    const float* __restrict__ src, unsigned short* __restrict__ dst, int Kr, int Nc)
{
  int i = blockIdx.x * 256 + threadIdx.x;
  if (i >= Kr * Nc) return;
  int k = i / Nc, n = i % Nc;
  dst[(size_t)n * Kr + k] = f2bf(src[(size_t)k * Nc + n]);
}

// causal depthwise conv (width 4) + SiLU on cols [0,DI) of C1 [MROWS,4096]
__global__ __launch_bounds__(256) void conv_silu_k(
    const float* __restrict__ C1, const float* __restrict__ Wc,
    float* __restrict__ uf, unsigned short* __restrict__ ub)
{
  int idx = blockIdx.x * 256 + threadIdx.x;     // r*DI + d
  int d = idx & (DI - 1);
  int r = idx >> 11;
  int l = r & (LSEQ - 1);
  float w0 = Wc[d * 4 + 0], w1 = Wc[d * 4 + 1], w2 = Wc[d * 4 + 2], w3 = Wc[d * 4 + 3];
  float v = C1[(size_t)r * 4096 + d] * w3;      // tap j=3 -> current sample
  if (l >= 1) v += C1[(size_t)(r - 1) * 4096 + d] * w2;
  if (l >= 2) v += C1[(size_t)(r - 2) * 4096 + d] * w1;
  if (l >= 3) v += C1[(size_t)(r - 3) * 4096 + d] * w0;
  float s = siluf(v);
  uf[idx] = s;
  ub[idx] = f2bf(s);
}

// x_dbl[:, 0:DTR] f32 -> bf16 [MROWS, DTR]
__global__ __launch_bounds__(256) void extract_dt_k(
    const float* __restrict__ xdbl, unsigned short* __restrict__ ddt)
{
  int idx = blockIdx.x * 256 + threadIdx.x;
  int r = idx >> 6, c = idx & 63;
  ddt[idx] = f2bf(xdbl[(size_t)r * 96 + c]);
}

// delta = softplus(delta + b_dt[d])  (in place)
__global__ __launch_bounds__(256) void bias_softplus_k(
    float* __restrict__ delta, const float* __restrict__ bdt)
{
  int idx = blockIdx.x * 256 + threadIdx.x;
  int d = idx & (DI - 1);
  float v = delta[idx] + bdt[d];
  delta[idx] = (v > 20.f) ? v : log1pf(__expf(v));
}

// selective scan: one thread per (stream,b,d); h[16] in registers.
// B/C chunks (64 steps x 32 f32, row stride 96) staged into LDS by the
// Tensor Data Mover (one 2D D# per chunk, issued by wave 0, TENSORcnt).
__global__ __launch_bounds__(256) void scan_k(
    const float* __restrict__ delta, const float* __restrict__ u,
    const float* __restrict__ xdbl,  const float* __restrict__ Alog,
    const float* __restrict__ Dv,    float* __restrict__ y)
{
  __shared__ float BC[64 * 32];                 // 64 timesteps x (B[16] | C[16])
  const int sb = blockIdx.x >> 3;               // (stream*B + b) in 0..3
  const int d  = ((blockIdx.x & 7) << 8) + threadIdx.x;
  const int rbase = sb * LSEQ;
  const unsigned bcLds = (unsigned)(uintptr_t)&BC[0];

  float Ar[DSTATE], h[DSTATE];
  #pragma unroll
  for (int n = 0; n < DSTATE; ++n) {
    Ar[n] = -__expf(Alog[(size_t)d * DSTATE + n]);
    h[n] = 0.f;
  }
  const float Dd = Dv[d];

  for (int lc = 0; lc < LSEQ; lc += 64) {
    __syncthreads();                            // previous chunk consumed
    if (threadIdx.x < 32) {
      // ---- Tensor DMA descriptor: 2D tile 32 x 64 f32, src stride 96 f32
      const unsigned long long ga =
          (unsigned long long)(uintptr_t)(xdbl + (size_t)(rbase + lc) * 96 + 64);
      u32x4 g0;
      g0[0] = 1u;                                   // count=1 (valid D#)
      g0[1] = bcLds;                                // lds_addr
      g0[2] = (unsigned)ga;                         // global_addr[31:0]
      g0[3] = (unsigned)(ga >> 32) | (2u << 30);    // global_addr[56:32] | type=2
      u32x8 g1;
      g1[0] = 2u << 16;                             // data_size = 4B
      g1[1] = 32u << 16;                            // tensor_dim0 = 32
      g1[2] = 64u << 16;                            // tensor_dim1 = 64
      g1[3] = 32u << 16;                            // tile_dim0 = 32
      g1[4] = 64u;                                  // tile_dim1 = 64
      g1[5] = 96u;                                  // tensor_dim0_stride = 96
      g1[6] = 0u;
      g1[7] = 0u;
      asm volatile("tensor_load_to_lds %0, %1" :: "s"(g0), "s"(g1) : "memory");
      __builtin_amdgcn_s_wait_tensorcnt(0);
    }
    __syncthreads();                            // chunk visible to all waves
    for (int li = 0; li < 64; ++li) {
      const size_t gi = (size_t)(rbase + lc + li) * DI + d;
      const float dl = delta[gi];
      const float ul = u[gi];
      const float du = dl * ul;
      float accv = 0.f;
      #pragma unroll
      for (int n = 0; n < DSTATE; ++n) {
        const float dA = __expf(dl * Ar[n]);
        h[n] = dA * h[n] + du * BC[li * 32 + n];
        accv += h[n] * BC[li * 32 + 16 + n];
      }
      y[gi] = accv + ul * Dd;
    }
  }
}

// gated = y * silu(res of the *other* stream), emitted as bf16 for the out-GEMM
__global__ __launch_bounds__(256) void gating_k(
    const float* __restrict__ y, const float* __restrict__ C1,
    unsigned short* __restrict__ gated)
{
  int idx = blockIdx.x * 256 + threadIdx.x;
  int r = idx >> 11, c = idx & (DI - 1);
  int rr = (r < 2048) ? r + 2048 : r - 2048;    // swap streams for the residual
  float res = C1[(size_t)rr * 4096 + 2048 + c];
  gated[idx] = f2bf(y[idx] * siluf(res));
}

// ---------------- host side ----------------
extern "C" void kernel_launch(void* const* d_in, const int* in_sizes, int n_in,
                              void* d_out, int out_size, void* d_ws, size_t ws_size,
                              hipStream_t stream) {
  (void)in_sizes; (void)n_in; (void)out_size; (void)ws_size;
  const float* x      = (const float*)d_in[0];
  const float* z      = (const float*)d_in[1];
  const float* W_in   = (const float*)d_in[2];
  const float* W_conv = (const float*)d_in[3];
  const float* W_xprj = (const float*)d_in[4];
  const float* W_dt   = (const float*)d_in[5];
  const float* b_dt   = (const float*)d_in[6];
  const float* A_log  = (const float*)d_in[7];
  const float* Dvec   = (const float*)d_in[8];
  const float* W_out  = (const float*)d_in[9];
  float* out = (float*)d_out;

  char* ws = (char*)d_ws;
  size_t off = 0;
  auto alloc = [&](size_t bytes) -> void* {
    void* p = (void*)(ws + off);
    off += (bytes + 255) & ~(size_t)255;
    return p;
  };
  unsigned short* XC    = (unsigned short*)alloc((size_t)MROWS * DM * 2);   // x;z bf16
  unsigned short* WINT  = (unsigned short*)alloc((size_t)4096 * 1024 * 2);  // W_in^T
  float*          C1    = (float*)alloc((size_t)MROWS * 4096 * 4);          // [u|res]
  float*          USF   = (float*)alloc((size_t)MROWS * DI * 4);            // conv+silu f32
  unsigned short* USB   = (unsigned short*)alloc((size_t)MROWS * DI * 2);   // bf16
  unsigned short* WXPT  = (unsigned short*)alloc((size_t)96 * DI * 2);      // W_xproj^T
  float*          XDBL  = (float*)alloc((size_t)MROWS * 96 * 4);
  unsigned short* DDT   = (unsigned short*)alloc((size_t)MROWS * DTR * 2);
  unsigned short* WDTT  = (unsigned short*)alloc((size_t)DI * DTR * 2);     // W_dt^T
  float*          DELTA = (float*)alloc((size_t)MROWS * DI * 4);
  float*          YB    = (float*)alloc((size_t)MROWS * DI * 4);
  unsigned short* GATED = (unsigned short*)alloc((size_t)MROWS * DI * 2);
  unsigned short* WOUTT = (unsigned short*)alloc((size_t)DM * DI * 2);      // W_out^T

  // 1) activations + weights -> bf16 (weights transposed so GEMM B is [N,K])
  cvt_f32_bf16_k<<<(2048 * DM) / 256, 256, 0, stream>>>(x, XC, 2048 * DM);
  cvt_f32_bf16_k<<<(2048 * DM) / 256, 256, 0, stream>>>(z, XC + (size_t)2048 * DM, 2048 * DM);
  tcvt_k<<<(1024 * 4096 + 255) / 256, 256, 0, stream>>>(W_in, WINT, 1024, 4096);

  // 2) C1 = [x;z] @ W_in   (4096x4096x1024)
  gemm_bf16_wmma<<<dim3(4096 / TN, MROWS / TM), 256, 0, stream>>>(
      XC, DM, WINT, DM, C1, 4096, MROWS, 4096, DM);

  // 3) causal dwconv + SiLU on the first DI columns
  conv_silu_k<<<(MROWS * DI) / 256, 256, 0, stream>>>(C1, W_conv, USF, USB);

  // 4) x_dbl = u @ W_xproj   (4096x96x2048)
  tcvt_k<<<(2048 * 96 + 255) / 256, 256, 0, stream>>>(W_xprj, WXPT, 2048, 96);
  gemm_bf16_wmma<<<dim3(1, MROWS / TM), 256, 0, stream>>>(
      USB, DI, WXPT, DI, XDBL, 96, MROWS, 96, DI);

  // 5) delta = softplus(x_dbl[:, :64] @ W_dt + b_dt)   (4096x2048x64)
  extract_dt_k<<<(MROWS * DTR) / 256, 256, 0, stream>>>(XDBL, DDT);
  tcvt_k<<<(64 * 2048 + 255) / 256, 256, 0, stream>>>(W_dt, WDTT, 64, 2048);
  gemm_bf16_wmma<<<dim3(DI / TN, MROWS / TM), 256, 0, stream>>>(
      DDT, DTR, WDTT, DTR, DELTA, DI, MROWS, DI, DTR);
  bias_softplus_k<<<(MROWS * DI) / 256, 256, 0, stream>>>(DELTA, b_dt);

  // 6) selective scan (both streams, both batches), B/C via TDM
  scan_k<<<32, 256, 0, stream>>>(DELTA, USF, XDBL, A_log, Dvec, YB);

  // 7) cross-stream gating -> bf16
  gating_k<<<(MROWS * DI) / 256, 256, 0, stream>>>(YB, C1, GATED);

  // 8) out GEMMs, row-permuted directly into d_out [B, 2L, DM]
  tcvt_k<<<(2048 * 1024 + 255) / 256, 256, 0, stream>>>(W_out, WOUTT, 2048, 1024);
  const int rowoff[4] = {0, 1024, 2048, 3072};   // gated rows: (img b0, img b1, aud b0, aud b1)
  const int outoff[4] = {0, 2048, 1024, 3072};   // d_out rows:  b*2L + (img?0:L)
  for (int i = 0; i < 4; ++i) {
    gemm_bf16_wmma<<<dim3(DM / TN, 1024 / TM), 256, 0, stream>>>(
        GATED + (size_t)rowoff[i] * DI, DI, WOUTT, DI,
        out + (size_t)outoff[i] * DM, DM, 1024, DM, DI);
  }
}
